// GSP_80023830659152
// MI455X (gfx1250) — matrix operand order repaired and checked by reading, not compiled, not run on hardware
//
#include <hip/hip_runtime.h>
#include <hip/hip_bf16.h>

// ---------------------------------------------------------------------------
// MI455X / gfx1250 implementation.
// Every einsum is a (34848 x 512) x (512 x 512) GEMM done with
// V_WMMA_F32_16X16X32_BF16 (bf16 inputs, fp32 accum). Weight tiles are
// async-staged into LDS (global_load_async_to_lds_b128 + s_wait_asynccnt),
// B fragments then come from LDS (ds_load_b128); A fragments are contiguous
// b128 global loads. Segment-max is a structured 4-neighbor gather.
// ---------------------------------------------------------------------------

typedef __attribute__((ext_vector_type(16))) __bf16 v16bf;
typedef __attribute__((ext_vector_type(8)))  float  v8f;

union Frag {
    uint4 q[2];
    v16bf v;
};

__device__ __forceinline__ v8f wmma_bf16(const Frag& a, const Frag& b, v8f c) {
    return __builtin_amdgcn_wmma_f32_16x16x32_bf16(
        false, a.v, false, b.v, (short)0, c, false, false);
}

// async copy of one 16B chunk per lane: global -> LDS, tracked by ASYNCcnt
__device__ __forceinline__ void async_g2l_b128(uint32_t lds_addr, const void* gaddr) {
    asm volatile("global_load_async_to_lds_b128 %0, %1, off"
                 :: "v"(lds_addr), "v"((uint64_t)(uintptr_t)gaddr)
                 : "memory");
}
__device__ __forceinline__ void wait_async0() {
    asm volatile("s_wait_asynccnt 0x0" ::: "memory");
}

#define GRIDW 33
#define NNODE 1089              // 33*33
#define BATCH 32
#define MROWS (BATCH * NNODE)   // 34848 (divisible by 16 and 32)
#define EMB 512
#define TOTE (MROWS * EMB)      // 17,842,176

// ------------------------- weight preparation ------------------------------

// plain cast (convg_W already is Wb[n=e][k=c])
__global__ void w_cast(const float* __restrict__ src, __hip_bfloat16* __restrict__ dst, int n) {
    int i = blockIdx.x * blockDim.x + threadIdx.x;
    if (i < n) dst[i] = __float2bfloat16(src[i]);
}

// transpose 4 stacked (512,512): Wb[l][f*512+e] = src[l][e*512+f]
__global__ void w_tr4(const float* __restrict__ src, __hip_bfloat16* __restrict__ dst) {
    int i = blockIdx.x * blockDim.x + threadIdx.x;   // 4*262144 total
    int l = i >> 18, rem = i & 262143;
    int e = rem >> 9, f = rem & 511;
    dst[l * 262144 + f * 512 + e] = __float2bfloat16(src[i]);
}

// conv2 head weights: (21,2560) -> 32x2560 bf16, rows 21..31 zero
__global__ void w_conv2(const float* __restrict__ src, __hip_bfloat16* __restrict__ dst) {
    int i = blockIdx.x * blockDim.x + threadIdx.x;   // 32*2560 total
    if (i >= 32 * 2560) return;
    int n = i / 2560, k = i - n * 2560;
    dst[i] = __float2bfloat16(n < 21 ? src[n * 2560 + k] : 0.0f);
}

// x (B,512,33,33) fp32 -> xt (M,512) bf16 row-major
__global__ void transpose_x(const float* __restrict__ x, __hip_bfloat16* __restrict__ xt) {
    int i = blockIdx.x * blockDim.x + threadIdx.x;
    if (i >= TOTE) return;
    int c = i & 511;
    int row = i >> 9;
    int b = row / NNODE, hw = row - b * NNODE;
    xt[i] = __float2bfloat16(x[((size_t)b * 512 + c) * NNODE + hw]);
}

// ------------------------------ main GEMM ----------------------------------
// D = act(A1*W1 (+ A2*W2) + bias). A row-major (M,512) bf16. W is N-major
// (Wb[n][k], 512x512) bf16. 8 waves: 4 along M x 2 along N, each 32x32 tile.
// Weight tile (64 rows x 512 K, 64KB per matrix) async-staged into LDS once.
template <int ACT, bool DUAL>
__global__ __launch_bounds__(256) void gemm_dual(
    const __hip_bfloat16* __restrict__ A1, const __hip_bfloat16* __restrict__ W1,
    const __hip_bfloat16* __restrict__ A2, const __hip_bfloat16* __restrict__ W2,
    const float* __restrict__ bias,
    float* __restrict__ outF, __hip_bfloat16* __restrict__ outB, int M)
{
    __shared__ __hip_bfloat16 sW[DUAL ? 2 : 1][64 * 512];

    // ---- async-stage weight tiles (ASYNCcnt path) ----
    {
        const char* src0 = (const char*)(W1 + (size_t)blockIdx.y * 64 * 512);
        const uint32_t l0 = (uint32_t)(uintptr_t)&sW[0][0];
        const uint32_t toff = (uint32_t)threadIdx.x * 16u;
#pragma unroll
        for (int i = 0; i < 16; ++i) {
            const uint32_t off = toff + (uint32_t)i * 4096u;
            async_g2l_b128(l0 + off, src0 + off);
        }
        if (DUAL) {
            const char* src1 = (const char*)(W2 + (size_t)blockIdx.y * 64 * 512);
            const uint32_t l1 = (uint32_t)(uintptr_t)&sW[DUAL ? 1 : 0][0];
#pragma unroll
            for (int i = 0; i < 16; ++i) {
                const uint32_t off = toff + (uint32_t)i * 4096u;
                async_g2l_b128(l1 + off, src1 + off);
            }
        }
        wait_async0();          // each wave drains its own async loads
        __syncthreads();        // tile visible to all waves
    }

    const int lane  = threadIdx.x & 31;
    const int w     = threadIdx.x >> 5;
    const int mBase = blockIdx.x * 128 + (w & 3) * 32;
    const int nBase = blockIdx.y * 64 + (w >> 2) * 32;
    const int half  = lane >> 4;     // 0: lanes 0-15, 1: lanes 16-31
    const int r16   = lane & 15;

    const int row0 = min(mBase + r16, M - 1);
    const int row1 = min(mBase + 16 + r16, M - 1);
    const int nl0  = (w >> 2) * 32 + r16;        // local W row in LDS
    const int nl1  = nl0 + 16;

    const uint4* a1p0 = reinterpret_cast<const uint4*>(A1) + (size_t)row0 * 64;
    const uint4* a1p1 = reinterpret_cast<const uint4*>(A1) + (size_t)row1 * 64;
    const uint4* w1p0 = reinterpret_cast<const uint4*>(&sW[0][0]) + (size_t)nl0 * 64;
    const uint4* w1p1 = reinterpret_cast<const uint4*>(&sW[0][0]) + (size_t)nl1 * 64;
    const uint4 *a2p0 = nullptr, *a2p1 = nullptr, *w2p0 = nullptr, *w2p1 = nullptr;
    if (DUAL) {
        a2p0 = reinterpret_cast<const uint4*>(A2) + (size_t)row0 * 64;
        a2p1 = reinterpret_cast<const uint4*>(A2) + (size_t)row1 * 64;
        w2p0 = reinterpret_cast<const uint4*>(&sW[DUAL ? 1 : 0][0]) + (size_t)nl0 * 64;
        w2p1 = reinterpret_cast<const uint4*>(&sW[DUAL ? 1 : 0][0]) + (size_t)nl1 * 64;
    }

    const v8f vz = {0.f, 0.f, 0.f, 0.f, 0.f, 0.f, 0.f, 0.f};
    v8f acc00 = vz, acc01 = vz, acc10 = vz, acc11 = vz;

#pragma unroll 4
    for (int k = 0; k < 512; k += 32) {
        // A fragment: lanes 0-15 hold K k..k+15, lanes 16-31 K k+16..k+31
        const int qa = (k >> 3) + 2 * half;
        // B fragment: lanes 0-15 hold K k..k+15, lanes 16-31 K k+16..k+31
        const int qb = (k >> 3) + 2 * half;

        Frag a0, a1, b0, b1;
        a0.q[0] = a1p0[qa];     a0.q[1] = a1p0[qa + 1];
        a1.q[0] = a1p1[qa];     a1.q[1] = a1p1[qa + 1];
        b0.q[0] = w1p0[qb];     b0.q[1] = w1p0[qb + 1];
        b1.q[0] = w1p1[qb];     b1.q[1] = w1p1[qb + 1];
        acc00 = wmma_bf16(a0, b0, acc00);
        acc01 = wmma_bf16(a0, b1, acc01);
        acc10 = wmma_bf16(a1, b0, acc10);
        acc11 = wmma_bf16(a1, b1, acc11);

        if (DUAL) {
            Frag c0, c1, d0, d1;
            c0.q[0] = a2p0[qa]; c0.q[1] = a2p0[qa + 1];
            c1.q[0] = a2p1[qa]; c1.q[1] = a2p1[qa + 1];
            d0.q[0] = w2p0[qb]; d0.q[1] = w2p0[qb + 1];
            d1.q[0] = w2p1[qb]; d1.q[1] = w2p1[qb + 1];
            acc00 = wmma_bf16(c0, d0, acc00);
            acc01 = wmma_bf16(c0, d1, acc01);
            acc10 = wmma_bf16(c1, d0, acc10);
            acc11 = wmma_bf16(c1, d1, acc11);
        }
    }

    // Epilogue. C/D layout: VGPR r -> M = r (+8 upper half); lane -> N.
    v8f accs[2][2] = {{acc00, acc01}, {acc10, acc11}};
#pragma unroll
    for (int nt = 0; nt < 2; ++nt) {
        const int n = nBase + nt * 16 + r16;
        const float bs = bias ? bias[n] : 0.0f;
#pragma unroll
        for (int mt = 0; mt < 2; ++mt) {
#pragma unroll
            for (int r = 0; r < 8; ++r) {
                const int row = mBase + mt * 16 + r + half * 8;
                if (row >= M) continue;
                float v = accs[mt][nt][r] + bs;
                if (ACT == 1) v = fmaxf(v, 0.0f);
                if (ACT == 2) v = 0.5f * v * (1.0f + erff(v * 0.70710678118654752f));
                const size_t o = (size_t)row * 512 + n;
                if (outF) outF[o] = v;
                if (outB) outB[o] = __float2bfloat16(v);
            }
        }
    }
}

// ------------------------------ batchnorm ----------------------------------

__global__ void zero_stats(float* s) {
    int i = threadIdx.x;
    if (i < 1024) s[i] = 0.0f;
}

__global__ __launch_bounds__(512) void bn_stats(const float* __restrict__ h,
                                                float* __restrict__ sums, int M) {
    int e = threadIdx.x;                       // 512 channels
    int rowsPer = (M + gridDim.x - 1) / gridDim.x;
    int r0 = blockIdx.x * rowsPer;
    int r1 = min(r0 + rowsPer, M);
    float s = 0.f, ss = 0.f;
    for (int r = r0; r < r1; ++r) {
        float v = h[(size_t)r * 512 + e];
        s += v; ss += v * v;
    }
    atomicAdd(&sums[e], s);
    atomicAdd(&sums[512 + e], ss);
}

__global__ void bn_final(const float* __restrict__ sums,
                         const float* __restrict__ gamma, const float* __restrict__ beta,
                         float* __restrict__ sc, float* __restrict__ sh, int M) {
    int e = threadIdx.x;
    float mean = sums[e] / (float)M;
    float var  = sums[512 + e] / (float)M - mean * mean;
    float s = gamma[e] * rsqrtf(var + 3.0e-4f);
    sc[e] = s;
    sh[e] = beta[e] - mean * s;
}

__global__ void bn_apply(const float* __restrict__ h, const float* __restrict__ sc,
                         const float* __restrict__ sh, __hip_bfloat16* __restrict__ g0) {
    int i = blockIdx.x * blockDim.x + threadIdx.x;
    if (i >= TOTE) return;
    int e = i & 511;
    float v = h[i] * sc[e] + sh[e];
    v = 0.5f * v * (1.0f + erff(v * 0.70710678118654752f));
    g0[i] = __float2bfloat16(v);
}

// -------------------------- neighbor max (SAGE) ----------------------------
// Incoming edges of node (i,j): (i,j-1), (i-1,j), (i-1,j-1), (i-1,j+1).
// m >= 0 after relu so init-0 max equals the reference's -inf->0 rule.
__global__ void nbr_max(const __hip_bfloat16* __restrict__ m, __hip_bfloat16* __restrict__ agg) {
    int idx = blockIdx.x * blockDim.x + threadIdx.x;
    if (idx >= TOTE) return;
    size_t base = (size_t)idx;
    int row = idx >> 9;
    int n = row % NNODE;
    int i = n / GRIDW, j = n - i * GRIDW;
    float v = 0.0f;
    if (j > 0) v = fmaxf(v, __bfloat162float(m[base - 512]));
    if (i > 0) {
        v = fmaxf(v, __bfloat162float(m[base - (size_t)GRIDW * 512]));
        if (j > 0)         v = fmaxf(v, __bfloat162float(m[base - (size_t)(GRIDW + 1) * 512]));
        if (j < GRIDW - 1) v = fmaxf(v, __bfloat162float(m[base - (size_t)(GRIDW - 1) * 512]));
    }
    agg[base] = __float2bfloat16(v);
}

// ------------------------------ final head ---------------------------------
// out[b,nc,hw] = sum_l feat_l[row,:] . Wb2[nc, l*512:...] + bias. One wave
// per 16 rows, two 16x16 n-tiles (NC=21 padded to 32 with zero weights).
struct F5 { const __hip_bfloat16* p[5]; };

__global__ __launch_bounds__(32) void final_conv(F5 feats, const __hip_bfloat16* __restrict__ Wb,
                                                 const float* __restrict__ bias,
                                                 float* __restrict__ out, int M) {
    const int lane = threadIdx.x & 31;
    const int half = lane >> 4;
    const int r16  = lane & 15;
    const int mBase = blockIdx.x * 16;
    const int rowA = min(mBase + r16, M - 1);

    const v8f vz = {0.f, 0.f, 0.f, 0.f, 0.f, 0.f, 0.f, 0.f};
    v8f acc[2] = {vz, vz};

    const uint4* pw = reinterpret_cast<const uint4*>(Wb);
#pragma unroll
    for (int l = 0; l < 5; ++l) {
        const uint4* pa = reinterpret_cast<const uint4*>(feats.p[l]) + (size_t)rowA * 64;
#pragma unroll 4
        for (int k = 0; k < 512; k += 32) {
            Frag a;
            a.q[0] = pa[(k >> 3) + 2 * half];
            a.q[1] = pa[(k >> 3) + 2 * half + 1];
            const int kk = l * 512 + k;
#pragma unroll
            for (int nt = 0; nt < 2; ++nt) {
                Frag b;
                const uint4* pr = pw + (size_t)(nt * 16 + r16) * 320;  // 2560/8
                b.q[0] = pr[(kk >> 3) + 2 * half];
                b.q[1] = pr[(kk >> 3) + 2 * half + 1];
                acc[nt] = wmma_bf16(a, b, acc[nt]);
            }
        }
    }

#pragma unroll
    for (int nt = 0; nt < 2; ++nt) {
        const int n = nt * 16 + r16;
        if (n >= 21) continue;
        const float bs = bias[n];
#pragma unroll
        for (int r = 0; r < 8; ++r) {
            const int row = mBase + r + half * 8;
            if (row >= M) continue;
            int b = row / NNODE, hw = row - b * NNODE;
            out[((size_t)b * 21 + n) * NNODE + hw] = acc[nt][r] + bs;
        }
    }
}

// ------------------------------- launcher ----------------------------------

extern "C" void kernel_launch(void* const* d_in, const int* in_sizes, int n_in,
                              void* d_out, int out_size, void* d_ws, size_t ws_size,
                              hipStream_t stream) {
    (void)in_sizes; (void)n_in; (void)out_size; (void)ws_size;

    const float* x        = (const float*)d_in[0];
    const float* convg_W  = (const float*)d_in[1];
    const float* convg_b  = (const float*)d_in[2];
    const float* bn_gamma = (const float*)d_in[3];
    const float* bn_beta  = (const float*)d_in[4];
    const float* s1_Wpool = (const float*)d_in[5];
    const float* s1_Wself = (const float*)d_in[6];
    const float* s1_Wneigh= (const float*)d_in[7];
    const float* s2_Wpool = (const float*)d_in[8];
    const float* s2_Wself = (const float*)d_in[9];
    const float* s2_Wneigh= (const float*)d_in[10];
    const float* lin_W    = (const float*)d_in[11];
    const float* s1_bpool = (const float*)d_in[12];
    const float* s1_b     = (const float*)d_in[13];
    const float* s2_bpool = (const float*)d_in[14];
    const float* s2_b     = (const float*)d_in[15];
    const float* lin_b    = (const float*)d_in[16];
    const float* conv2_W  = (const float*)d_in[17];
    const float* conv2_b  = (const float*)d_in[18];
    // d_in[19]/d_in[20] (src,dst) unused: grid adjacency is computed analytically.

    const int M = MROWS;
    const size_t SZ_W   = (size_t)512 * 512 * sizeof(__hip_bfloat16);   // 512 KB
    const size_t SZ_ACT = (size_t)M * 512 * sizeof(__hip_bfloat16);     // ~35.7 MB

    char* p = (char*)d_ws;
    auto take = [&](size_t bytes) { char* r = p; p += (bytes + 255) & ~(size_t)255; return r; };

    __hip_bfloat16* wb_convg  = (__hip_bfloat16*)take(SZ_W);
    __hip_bfloat16* wb_s1pool = (__hip_bfloat16*)take(4 * SZ_W);
    __hip_bfloat16* wb_s1self = (__hip_bfloat16*)take(4 * SZ_W);
    __hip_bfloat16* wb_s1ngh  = (__hip_bfloat16*)take(4 * SZ_W);
    __hip_bfloat16* wb_s2pool = (__hip_bfloat16*)take(4 * SZ_W);
    __hip_bfloat16* wb_s2self = (__hip_bfloat16*)take(4 * SZ_W);
    __hip_bfloat16* wb_s2ngh  = (__hip_bfloat16*)take(4 * SZ_W);
    __hip_bfloat16* wb_lin    = (__hip_bfloat16*)take(4 * SZ_W);
    __hip_bfloat16* wb_conv2  = (__hip_bfloat16*)take((size_t)32 * 2560 * 2);
    float* stats    = (float*)take(1024 * sizeof(float));
    float* bn_scale = (float*)take(512 * sizeof(float));
    float* bn_shift = (float*)take(512 * sizeof(float));
    __hip_bfloat16* featsBase = (__hip_bfloat16*)take(5 * SZ_ACT);
    char* regionA = take(2 * SZ_ACT);   // xt aliases {m, agg} (disjoint in time)
    char* regionB = take(2 * SZ_ACT);   // h(fp32) aliases {tmpA, tmpB}

    __hip_bfloat16* xt    = (__hip_bfloat16*)regionA;
    __hip_bfloat16* m_b   = (__hip_bfloat16*)regionA;
    __hip_bfloat16* agg_b = (__hip_bfloat16*)(regionA + SZ_ACT);
    float*          h     = (float*)regionB;                  // M*512*4 == 2*SZ_ACT
    __hip_bfloat16* tmpA  = (__hip_bfloat16*)regionB;
    __hip_bfloat16* tmpB  = (__hip_bfloat16*)(regionB + SZ_ACT);

    auto feat = [&](int l) { return featsBase + (size_t)l * M * 512; };

    // ---- weight prep (bf16, N-major) ----
    w_cast<<<262144 / 256, 256, 0, stream>>>(convg_W, wb_convg, 262144);
    w_tr4<<<4096, 256, 0, stream>>>(s1_Wpool,  wb_s1pool);
    w_tr4<<<4096, 256, 0, stream>>>(s1_Wself,  wb_s1self);
    w_tr4<<<4096, 256, 0, stream>>>(s1_Wneigh, wb_s1ngh);
    w_tr4<<<4096, 256, 0, stream>>>(s2_Wpool,  wb_s2pool);
    w_tr4<<<4096, 256, 0, stream>>>(s2_Wself,  wb_s2self);
    w_tr4<<<4096, 256, 0, stream>>>(s2_Wneigh, wb_s2ngh);
    w_tr4<<<4096, 256, 0, stream>>>(lin_W,     wb_lin);
    w_conv2<<<(32 * 2560 + 255) / 256, 256, 0, stream>>>(conv2_W, wb_conv2);

    // ---- input transpose + 1x1 conv GEMM (fp32 out for BN stats) ----
    transpose_x<<<TOTE / 256, 256, 0, stream>>>(x, xt);
    dim3 gg((M + 127) / 128, 8), gb(256);
    gemm_dual<0, false><<<gg, gb, 0, stream>>>(xt, wb_convg, nullptr, nullptr,
                                               convg_b, h, nullptr, M);

    // ---- batchnorm + gelu -> feats[0] ----
    zero_stats<<<1, 1024, 0, stream>>>(stats);
    bn_stats<<<64, 512, 0, stream>>>(h, stats, M);
    bn_final<<<1, 512, 0, stream>>>(stats, bn_gamma, bn_beta, bn_scale, bn_shift, M);
    bn_apply<<<TOTE / 256, 256, 0, stream>>>(h, bn_scale, bn_shift, feat(0));

    // ---- 4 GNN layers ----
    for (int i = 0; i < 4; ++i) {
        const size_t wo = (size_t)i * 262144;
        const __hip_bfloat16* gin = feat(i);
        // SAGE 1
        gemm_dual<1, false><<<gg, gb, 0, stream>>>(gin, wb_s1pool + wo, nullptr, nullptr,
                                                   s1_bpool + i * 512, nullptr, m_b, M);
        nbr_max<<<TOTE / 256, 256, 0, stream>>>(m_b, agg_b);
        gemm_dual<2, true><<<gg, gb, 0, stream>>>(gin, wb_s1self + wo, agg_b, wb_s1ngh + wo,
                                                  s1_b + i * 512, nullptr, tmpA, M);
        // SAGE 2
        gemm_dual<1, false><<<gg, gb, 0, stream>>>(tmpA, wb_s2pool + wo, nullptr, nullptr,
                                                   s2_bpool + i * 512, nullptr, m_b, M);
        nbr_max<<<TOTE / 256, 256, 0, stream>>>(m_b, agg_b);
        gemm_dual<2, true><<<gg, gb, 0, stream>>>(tmpA, wb_s2self + wo, agg_b, wb_s2ngh + wo,
                                                  s2_b + i * 512, nullptr, tmpB, M);
        // lin -> feats[i+1]
        gemm_dual<2, false><<<gg, gb, 0, stream>>>(tmpB, wb_lin + wo, nullptr, nullptr,
                                                   lin_b + i * 512, nullptr, feat(i + 1), M);
    }

    // ---- final 21-class head over concatenated features ----
    F5 f5;
    for (int l = 0; l < 5; ++l) f5.p[l] = feat(l);
    final_conv<<<M / 16, 32, 0, stream>>>(f5, wb_conv2, conv2_b, (float*)d_out, M);
}
